// CustomLoss_38190849196640
// MI455X (gfx1250) — compile-verified, matrix operand order checked by reading, not run on hardware
//
#include <hip/hip_runtime.h>

// ---------------------------------------------------------------------------
// Gaussian W2 loss via Newton-Schulz matrix square roots (GEMM-only, WMMA bf16)
// ---------------------------------------------------------------------------
#define NMAT   2048
#define TILE   128
#define KTILE  32
#define LDSROW 40          // padded LDS row stride (elements): 80B, 16B-aligned
#define NS_ITERS 18

typedef __attribute__((ext_vector_type(8)))  float  v8f;
typedef __attribute__((ext_vector_type(2)))  __bf16 v2bf;
typedef __attribute__((ext_vector_type(8)))  __bf16 v8bf;
typedef __attribute__((ext_vector_type(16))) __bf16 v16bf;

__device__ __forceinline__ v16bf load_frag16(const __bf16* p) {
  v8bf lo = *(const v8bf*)(p);       // 16B aligned chunk, K = kbase..kbase+7
  v8bf hi = *(const v8bf*)(p + 16);  // 16B aligned chunk, K = kbase+16..+23
  v16bf r;
#pragma unroll
  for (int i = 0; i < 8; ++i) { r[i] = lo[i]; r[i + 8] = hi[i]; }
  return r;
}

// C = scale * (A @ B) + diagAdd * I     (A,B,C fp32 NMATxNMAT row-major)
// 256 threads = 8 waves; block tile 128x128; double-buffered LDS, one barrier
// per K-step; each wave: 2x4 grid of 16x16 WMMA tiles (8 v_wmma per K-step).
__global__ __launch_bounds__(256) void gemm_bf16_wmma(
    const float* __restrict__ A, const float* __restrict__ B,
    float* __restrict__ C, float scale, float diagAdd)
{
  __shared__ __bf16 As[2][TILE * LDSROW];   // As[buf][m][k]
  __shared__ __bf16 Bs[2][TILE * LDSROW];   // Bs[buf][n][k]  (transposed B tile)

  const int tid  = threadIdx.x;
  const int bm   = blockIdx.y * TILE;
  const int bn   = blockIdx.x * TILE;

  const int wave = tid >> 5;
  const int lane = tid & 31;
  const int half = lane >> 4;     // 0: lanes 0-15, 1: lanes 16-31
  const int lid  = lane & 15;
  const int wm   = wave & 3;      // 4 wave rows  -> 32 rows each
  const int wn   = wave >> 2;     // 2 wave cols  -> 64 cols each
  const int kbase = half * 8;     // element offset of first 8-wide K chunk

  v8f acc[2][4];
#pragma unroll
  for (int tm = 0; tm < 2; ++tm)
#pragma unroll
    for (int tn = 0; tn < 4; ++tn)
#pragma unroll
      for (int r = 0; r < 8; ++r) acc[tm][tn][r] = 0.0f;

  // cooperative load mappings
  const int a_row = tid >> 1;            // 0..127
  const int a_col = (tid & 1) << 4;      // 0 or 16
  const int b_k   = (tid & 15) << 1;     // even k: 0,2,..,30 (pair of K rows)
  const int b_nc  = (tid >> 4) << 3;     // 0,8,..,120 (8 columns per thread)

  const float* aP  = A + (size_t)(bm + a_row) * NMAT + a_col;
  const float* bP0 = B + (size_t)b_k * NMAT + bn + b_nc;
  const float* bP1 = bP0 + NMAT;

  float4 a0, a1, a2, a3;   // A: 16 floats (one row chunk)
  float4 r0a, r0b, r1a, r1b; // B: 8 floats of row k, 8 floats of row k+1

  auto load_regs = [&](int k0) {
    const float4* ag = (const float4*)(aP + k0);
    a0 = ag[0]; a1 = ag[1]; a2 = ag[2]; a3 = ag[3];
    const float4* bg0 = (const float4*)(bP0 + (size_t)k0 * NMAT);
    r0a = bg0[0]; r0b = bg0[1];
    const float4* bg1 = (const float4*)(bP1 + (size_t)k0 * NMAT);
    r1a = bg1[0]; r1b = bg1[1];
  };

  auto store_tile = [&](int buf) {
    // A: 16 contiguous bf16 -> two packed 16B LDS stores
    __bf16* as = &As[buf][a_row * LDSROW + a_col];
    v8bf lo, hi;
    lo[0] = (__bf16)a0.x; lo[1] = (__bf16)a0.y; lo[2] = (__bf16)a0.z; lo[3] = (__bf16)a0.w;
    lo[4] = (__bf16)a1.x; lo[5] = (__bf16)a1.y; lo[6] = (__bf16)a1.z; lo[7] = (__bf16)a1.w;
    hi[0] = (__bf16)a2.x; hi[1] = (__bf16)a2.y; hi[2] = (__bf16)a2.z; hi[3] = (__bf16)a2.w;
    hi[4] = (__bf16)a3.x; hi[5] = (__bf16)a3.y; hi[6] = (__bf16)a3.z; hi[7] = (__bf16)a3.w;
    *(v8bf*)as = lo;
    *(v8bf*)(as + 8) = hi;
    // B transpose: pack (k, k+1) pair per column -> 8 b32 stores, conflict-free
    float c0[8] = {r0a.x, r0a.y, r0a.z, r0a.w, r0b.x, r0b.y, r0b.z, r0b.w};
    float c1[8] = {r1a.x, r1a.y, r1a.z, r1a.w, r1b.x, r1b.y, r1b.z, r1b.w};
#pragma unroll
    for (int i = 0; i < 8; ++i) {
      v2bf p; p[0] = (__bf16)c0[i]; p[1] = (__bf16)c1[i];
      *(v2bf*)&Bs[buf][(b_nc + i) * LDSROW + b_k] = p;
    }
  };

  // prologue: stage tile 0
  load_regs(0);
  store_tile(0);

  int cur = 0;
  for (int k0 = 0; k0 < NMAT; k0 += KTILE) {
    __syncthreads();
    const bool more = (k0 + KTILE) < NMAT;
    if (more) load_regs(k0 + KTILE);   // global prefetch overlaps WMMAs below

    // ---- fragments (per ISA 16-bit A/B striping) ----
    v16bf afrag[2], bfrag[4];
#pragma unroll
    for (int tm = 0; tm < 2; ++tm) {
      const int row = wm * 32 + tm * 16 + lid;
      afrag[tm] = load_frag16(&As[cur][row * LDSROW + kbase]);
    }
#pragma unroll
    for (int tn = 0; tn < 4; ++tn) {
      const int col = wn * 64 + tn * 16 + lid;
      bfrag[tn] = load_frag16(&Bs[cur][col * LDSROW + kbase]);
    }

#pragma unroll
    for (int tm = 0; tm < 2; ++tm)
#pragma unroll
      for (int tn = 0; tn < 4; ++tn)
        acc[tm][tn] = __builtin_amdgcn_wmma_f32_16x16x32_bf16(
            false, afrag[tm], false, bfrag[tn],
            (short)0, acc[tm][tn], false, false);

    if (more) store_tile(cur ^ 1);     // safe: buf^1 readers finished pre-barrier
    cur ^= 1;
  }

  // ---- epilogue: C = scale*acc + diagAdd*I ----
#pragma unroll
  for (int tm = 0; tm < 2; ++tm)
#pragma unroll
    for (int tn = 0; tn < 4; ++tn) {
      const int gcol = bn + wn * 64 + tn * 16 + lid;
#pragma unroll
      for (int r = 0; r < 8; ++r) {
        const int grow = bm + wm * 32 + tm * 16 + half * 8 + r;
        float v = scale * acc[tm][tn][r];
        if (grow == gcol) v += diagAdd;
        C[(size_t)grow * NMAT + gcol] = v;
      }
    }
}

// ---------------------------------------------------------------------------
// O(N^2) helper kernels
// ---------------------------------------------------------------------------
__global__ void unpack_cov_kernel(const float* __restrict__ tri,
                                  float* __restrict__ C) {
  const int idx = blockIdx.x * blockDim.x + threadIdx.x;   // exact grid
  const int i = idx >> 11, j = idx & 2047;
  const int a = i < j ? i : j;
  const int b = i < j ? j : i;
  // upper-triangle row-major offset of (a,b), a<=b
  const size_t off = (size_t)a * NMAT - (size_t)(a * (a - 1)) / 2 + (size_t)(b - a);
  C[idx] = tri[off];
}

__global__ void identity_kernel(float* __restrict__ Z) {
  const int idx = blockIdx.x * blockDim.x + threadIdx.x;
  Z[idx] = ((idx >> 11) == (idx & 2047)) ? 1.0f : 0.0f;
}

// Y = A * rsqrt(S[0])   (divide by Frobenius norm; S holds ||A||_F^2)
__global__ void normalize_kernel(const float* __restrict__ A,
                                 float* __restrict__ Y,
                                 const float* __restrict__ Sslot) {
  const int idx = blockIdx.x * blockDim.x + threadIdx.x;
  Y[idx] = A[idx] * rsqrtf(Sslot[0]);
}

__global__ void mu_diff_kernel(const float* __restrict__ mp,
                               const float* __restrict__ mt,
                               float* __restrict__ Sslot) {
  __shared__ float sm[256];
  const int tid = threadIdx.x;
  float s = 0.0f;
  for (int i = tid; i < NMAT; i += 256) { float d = mp[i] - mt[i]; s += d * d; }
  sm[tid] = s; __syncthreads();
  for (int o = 128; o; o >>= 1) { if (tid < o) sm[tid] += sm[tid + o]; __syncthreads(); }
  if (tid == 0) Sslot[0] = sm[0];
}

__global__ void trace_kernel(const float* __restrict__ M,
                             float* __restrict__ Sslot) {
  __shared__ float sm[256];
  const int tid = threadIdx.x;
  float s = 0.0f;
  for (int i = tid; i < NMAT; i += 256) s += M[(size_t)i * (NMAT + 1)];
  sm[tid] = s; __syncthreads();
  for (int o = 128; o; o >>= 1) { if (tid < o) sm[tid] += sm[tid + o]; __syncthreads(); }
  if (tid == 0) Sslot[0] = sm[0];
}

__global__ void frob_partial_kernel(const float* __restrict__ M,
                                    float* __restrict__ partials) {
  __shared__ float sm[256];
  const int tid = threadIdx.x;
  const size_t n = (size_t)NMAT * NMAT;
  float s = 0.0f;
  for (size_t i = (size_t)blockIdx.x * 256 + tid; i < n; i += 256 * 256) {
    float v = M[i]; s += v * v;
  }
  sm[tid] = s; __syncthreads();
  for (int o = 128; o; o >>= 1) { if (tid < o) sm[tid] += sm[tid + o]; __syncthreads(); }
  if (tid == 0) partials[blockIdx.x] = sm[0];
}

__global__ void sum256_kernel(const float* __restrict__ partials,
                              float* __restrict__ Sslot) {
  __shared__ float sm[256];
  const int tid = threadIdx.x;
  sm[tid] = partials[tid]; __syncthreads();
  for (int o = 128; o; o >>= 1) { if (tid < o) sm[tid] += sm[tid + o]; __syncthreads(); }
  if (tid == 0) Sslot[0] = sm[0];
}

// out = S0/D + S1 + S2 + 2 * S3^(1/4) * S4^(1/4) * S5
__global__ void combine_kernel(const float* __restrict__ S,
                               float* __restrict__ out) {
  if (threadIdx.x == 0 && blockIdx.x == 0) {
    const float q = sqrtf(sqrtf(S[3])) * sqrtf(sqrtf(S[4])) * S[5];
    out[0] = S[0] * (1.0f / NMAT) + S[1] + S[2] + 2.0f * q;
  }
}

// ---------------------------------------------------------------------------
// Host orchestration (capture-safe: launches only, fixed iteration counts)
// ---------------------------------------------------------------------------
extern "C" void kernel_launch(void* const* d_in, const int* in_sizes, int n_in,
                              void* d_out, int out_size, void* d_ws, size_t ws_size,
                              hipStream_t stream) {
  const float* pred = (const float*)d_in[0];
  const float* targ = (const float*)d_in[1];
  const float* mu_p = pred;           const float* tri_p = pred + NMAT;
  const float* mu_t = targ;           const float* tri_t = targ + NMAT;

  float* S        = (float*)d_ws;                    // 64 scalar slots
  float* partials = S + 64;                          // 256 partials
  float* base     = (float*)((char*)d_ws + 4096);
  const size_t MS = (size_t)NMAT * NMAT;
  float* M0 = base + 0 * MS;   // Cp, then NS temp T
  float* M1 = base + 1 * MS;   // Ct, then Z2
  float* M2 = base + 2 * MS;
  float* M3 = base + 3 * MS;
  float* M4 = base + 4 * MS;
  float* M5 = base + 5 * MS;   // total scratch ~101 MB

  const int  EW_BLOCKS = (int)(MS / 256);            // 16384, exact cover
  const dim3 gg(NMAT / TILE, NMAT / TILE), gb(256);

  // unpack + scalar terms
  unpack_cov_kernel<<<EW_BLOCKS, 256, 0, stream>>>(tri_p, M0);
  unpack_cov_kernel<<<EW_BLOCKS, 256, 0, stream>>>(tri_t, M1);
  mu_diff_kernel<<<1, 256, 0, stream>>>(mu_p, mu_t, S + 0);
  trace_kernel<<<1, 256, 0, stream>>>(M0, S + 1);
  trace_kernel<<<1, 256, 0, stream>>>(M1, S + 2);
  frob_partial_kernel<<<256, 256, 0, stream>>>(M0, partials);
  sum256_kernel<<<1, 256, 0, stream>>>(partials, S + 3);

  // Newton-Schulz #1: sqrtm(Cp / ||Cp||_F)
  normalize_kernel<<<EW_BLOCKS, 256, 0, stream>>>(M0, M2, S + 3);  // Y0
  identity_kernel<<<EW_BLOCKS, 256, 0, stream>>>(M3);              // Z0
  float *Y = M2, *Z = M3, *Yn = M4, *Zn = M5, *T = M0, *tmpp;
  for (int it = 0; it < NS_ITERS; ++it) {
    gemm_bf16_wmma<<<gg, gb, 0, stream>>>(Z, Y, T, -1.0f, 3.0f);   // T = 3I - Z@Y
    gemm_bf16_wmma<<<gg, gb, 0, stream>>>(Y, T, Yn, 0.5f, 0.0f);   // Y' = .5 Y@T
    gemm_bf16_wmma<<<gg, gb, 0, stream>>>(T, Z, Zn, 0.5f, 0.0f);   // Z' = .5 T@Z
    tmpp = Y; Y = Yn; Yn = tmpp;
    tmpp = Z; Z = Zn; Zn = tmpp;
  }

  // X = Y @ Ct @ Y   (M = ||Cp||_F * X)
  gemm_bf16_wmma<<<gg, gb, 0, stream>>>(Y, M1, T, 1.0f, 0.0f);     // T = Y@Ct
  float* X = Zn;                                                   // free buffer
  gemm_bf16_wmma<<<gg, gb, 0, stream>>>(T, Y, X, 1.0f, 0.0f);
  frob_partial_kernel<<<256, 256, 0, stream>>>(X, partials);
  sum256_kernel<<<1, 256, 0, stream>>>(partials, S + 4);

  // Newton-Schulz #2: sqrtm(X / ||X||_F); we only need trace(Y_final)
  float *Y2 = Yn, *Z2 = M1, *Yn2 = Y, *Zn2 = X, *T2 = T;           // 5 distinct
  normalize_kernel<<<EW_BLOCKS, 256, 0, stream>>>(X, Y2, S + 4);   // (X read before overwrite as Zn2)
  identity_kernel<<<EW_BLOCKS, 256, 0, stream>>>(Z2);
  for (int it = 0; it < NS_ITERS; ++it) {
    gemm_bf16_wmma<<<gg, gb, 0, stream>>>(Z2, Y2, T2, -1.0f, 3.0f);
    gemm_bf16_wmma<<<gg, gb, 0, stream>>>(Y2, T2, Yn2, 0.5f, 0.0f);
    gemm_bf16_wmma<<<gg, gb, 0, stream>>>(T2, Z2, Zn2, 0.5f, 0.0f);
    tmpp = Y2; Y2 = Yn2; Yn2 = tmpp;
    tmpp = Z2; Z2 = Zn2; Zn2 = tmpp;
  }
  trace_kernel<<<1, 256, 0, stream>>>(Y2, S + 5);

  combine_kernel<<<1, 64, 0, stream>>>(S, (float*)d_out);
}
